// GRUCell_86964497809430
// MI455X (gfx1250) — matrix operand order repaired and checked by reading, not compiled, not run on hardware
//
#include <hip/hip_runtime.h>

// ---------------------------------------------------------------------------
// Types for CDNA5 WMMA
// ---------------------------------------------------------------------------
typedef __attribute__((ext_vector_type(16))) __bf16 v16bf;
typedef __attribute__((ext_vector_type(8)))  float  v8f;

union Frag16 {
    v16bf v;
    uint4 q[2];
};

#define EPS 1e-5f
#define MAXN (1.0f - 1e-5f)

// ---------------------------------------------------------------------------
// Helpers
// ---------------------------------------------------------------------------
__device__ __forceinline__ unsigned short f2bf(float f) {
    unsigned int u = __float_as_uint(f);
    unsigned int rnd = 0x7FFFu + ((u >> 16) & 1u);
    return (unsigned short)((u + rnd) >> 16);
}

__device__ __forceinline__ float clamp01(float v) {
    return fminf(fmaxf(v, EPS), MAXN);
}

__device__ __forceinline__ float dot4(const float* a, const float* b) {
    return a[0]*b[0] + a[1]*b[1] + a[2]*b[2] + a[3]*b[3];
}

// Mobius addition on one ball (c = 1)
__device__ __forceinline__ void mob_add4(const float* u, const float* v, float* o) {
    float uv = dot4(u, v), uu = dot4(u, u), vv = dot4(v, v);
    float cu = 1.0f + 2.0f*uv + vv;
    float cv = 1.0f - uu;
    float den = 1.0f + 2.0f*uv + uu*vv;
    den = fmaxf(den, EPS);
    float inv = 1.0f / den;
#pragma unroll
    for (int i = 0; i < 4; ++i) o[i] = (cu*u[i] + cv*v[i]) * inv;
}

// log map at origin, per ball
__device__ __forceinline__ void logmap4(const float* y, float* o) {
    float n = sqrtf(dot4(y, y));
    float nc = clamp01(n);
    float s = atanhf(nc) / nc;
#pragma unroll
    for (int i = 0; i < 4; ++i) o[i] = s * y[i];
}

// diagonal Mobius scaling: hyperbolic x gated by euclidean u, per ball
__device__ __forceinline__ void mob_pp4(const float* x, const float* u, float* o) {
    float ux[4];
#pragma unroll
    for (int i = 0; i < 4; ++i) ux[i] = u[i] * x[i];
    float xn = clamp01(sqrtf(dot4(x, x)));
    float uxn = fmaxf(sqrtf(dot4(ux, ux)), EPS);
    float s = tanhf(uxn / xn * atanhf(xn)) / uxn;
#pragma unroll
    for (int i = 0; i < 4; ++i) o[i] = s * ux[i];
}

__device__ __forceinline__ float block_sum256(float v, float* red, int tid) {
    red[tid] = v;
    __syncthreads();
#pragma unroll
    for (int s = 128; s > 0; s >>= 1) {
        if (tid < s) red[tid] += red[tid + s];
        __syncthreads();
    }
    float r = red[0];
    __syncthreads();
    return r;
}

// CDNA5 async global->LDS copy, tracked by ASYNCcnt
__device__ __forceinline__ void async_copy_b128(unsigned lds_off, unsigned long long gaddr) {
    asm volatile("global_load_async_to_lds_b128 %0, %1, off"
                 :: "v"(lds_off), "v"(gaddr)
                 : "memory");
}

__device__ __forceinline__ void wait_async0() {
    asm volatile("s_wait_asynccnt 0x0" ::: "memory");
}

// ---------------------------------------------------------------------------
// f32 -> bf16 conversion (row-major passthrough, for activations)
// ---------------------------------------------------------------------------
__global__ __launch_bounds__(256) void cvt_bf16(const float* __restrict__ in,
                                                unsigned short* __restrict__ out,
                                                int n) {
    int i = (blockIdx.x * 256 + threadIdx.x) * 4;
    if (i >= n) return;
    float4 v = *(const float4*)&in[i];
    out[i + 0] = f2bf(v.x);
    out[i + 1] = f2bf(v.y);
    out[i + 2] = f2bf(v.z);
    out[i + 3] = f2bf(v.w);
}

// ---------------------------------------------------------------------------
// f32 (K,N) -> bf16 transposed (N,K): weights converted once so GEMM tiles
// are contiguous on both operands (no per-tile LDS scatter).
// ---------------------------------------------------------------------------
__global__ __launch_bounds__(256) void cvt_bf16_T(const float* __restrict__ in,
                                                  unsigned short* __restrict__ out,
                                                  int K, int N) {
    __shared__ float tile[32][33];
    const int bn = blockIdx.x * 32;   // N offset
    const int bk = blockIdx.y * 32;   // K offset
    const int tx = threadIdx.x & 31;
    const int ty = threadIdx.x >> 5;  // 0..7
#pragma unroll
    for (int i = ty; i < 32; i += 8)
        tile[i][tx] = in[(size_t)(bk + i) * N + bn + tx];
    __syncthreads();
#pragma unroll
    for (int i = ty; i < 32; i += 8)
        out[(size_t)(bn + i) * K + bk + tx] = f2bf(tile[tx][i]);
}

// ---------------------------------------------------------------------------
// bf16 GEMM via V_WMMA_F32_16X16X32_BF16, async double-buffered LDS tiles.
// C(M,N) f32 = A(M,K) bf16 @ Wt(N,K)^T bf16.  M%256==0, N%128==0, K%32==0.
// 256 threads = 8 waves; block tile 256x128; wave tile 64x64
// -> 16 WMMAs per wave per K-step against 16 ds_load_b128 (1.0 loads/WMMA)
// ---------------------------------------------------------------------------
#define TM 256
#define TN 128
#define TK 32

__global__ __launch_bounds__(256) void gemm_bf16(const unsigned short* __restrict__ A,
                                                 const unsigned short* __restrict__ Wt,
                                                 float* __restrict__ C,
                                                 int M, int N, int K) {
    __shared__ __align__(16) unsigned short ldsA[2][TM * TK];  // [m][k]  16KB/buf
    __shared__ __align__(16) unsigned short ldsB[2][TN * TK];  // [n][k]   8KB/buf

    const int tid  = threadIdx.x;
    const int lane = tid & 31;
    const int wv   = tid >> 5;
    const int wy   = wv >> 1;  // 0..3 : 64-row strip
    const int wx   = wv & 1;   // 0..1 : 64-col strip
    const size_t rowBase = (size_t)blockIdx.y * TM;
    const size_t colBase = (size_t)blockIdx.x * TN;

    v8f acc[4][4] = {};

    const int m   = lane & 15;
    const int hi  = (lane & 16) ? 8 : 0;  // A-fragment K interleave (ISA layout)
    const int klo = (lane & 16);          // unused; B loads use hi, same k order as A

    // async tile fill: A = 4 chunks, B = 2 chunks of 16B per thread per K-step
    auto load_tiles = [&](int k0, int buf) {
#pragma unroll
        for (int j = 0; j < 4; ++j) {
            const int c  = tid + 256 * j;     // 0..1023
            const int r  = c >> 2;            // 0..255
            const int kc = (c & 3) * 8;       // 0,8,16,24
            async_copy_b128((unsigned)(size_t)&ldsA[buf][r * TK + kc],
                            (unsigned long long)&A[(rowBase + r) * K + k0 + kc]);
        }
#pragma unroll
        for (int j = 0; j < 2; ++j) {
            const int c  = tid + 256 * j;     // 0..511
            const int r  = c >> 2;            // 0..127
            const int kc = (c & 3) * 8;
            async_copy_b128((unsigned)(size_t)&ldsB[buf][r * TK + kc],
                            (unsigned long long)&Wt[(colBase + r) * K + k0 + kc]);
        }
    };

    load_tiles(0, 0);

    const int nIter = K / TK;
    for (int it = 0; it < nIter; ++it) {
        const int buf = it & 1;
        wait_async0();        // my fills of buf done
        __syncthreads();      // everyone's fills done + everyone done reading buf^1
        if (it + 1 < nIter) load_tiles((it + 1) * TK, buf ^ 1);

        Frag16 fa[4], fb[4];
#pragma unroll
        for (int s = 0; s < 4; ++s) {
            const unsigned short* p = &ldsA[buf][(wy * 64 + s * 16 + m) * TK];
            fa[s].q[0] = *(const uint4*)&p[hi];
            fa[s].q[1] = *(const uint4*)&p[16 + hi];
        }
#pragma unroll
        for (int j = 0; j < 4; ++j) {
            const unsigned short* p = &ldsB[buf][(wx * 64 + j * 16 + m) * TK];
            fb[j].q[0] = *(const uint4*)&p[hi];
            fb[j].q[1] = *(const uint4*)&p[16 + hi];
        }
#pragma unroll
        for (int s = 0; s < 4; ++s)
#pragma unroll
            for (int j = 0; j < 4; ++j)
                acc[s][j] = __builtin_amdgcn_wmma_f32_16x16x32_bf16(
                    false, fa[s].v, false, fb[j].v, (short)0, acc[s][j], false, false);
        // no trailing barrier needed: next writes target buf^1, gated by the
        // top-of-iteration wait+barrier.
    }

    // C/D layout: VGPR r -> M = r + (lane>=16 ? 8 : 0), N = lane&15
#pragma unroll
    for (int s = 0; s < 4; ++s) {
        const size_t orow = rowBase + wy * 64 + s * 16 + ((lane & 16) ? 8 : 0);
#pragma unroll
        for (int j = 0; j < 4; ++j) {
            const size_t ocol = colBase + wx * 64 + j * 16 + (lane & 15);
#pragma unroll
            for (int r = 0; r < 8; ++r)
                C[(orow + r) * N + ocol] = acc[s][j][r];
        }
    }
}

// ---------------------------------------------------------------------------
// Elementwise phase 1: gates z and r, r (.) h, scaled Ur@x, per batch row.
// One block (256 threads) per row; thread t owns ball t (4 elements).
// ---------------------------------------------------------------------------
__global__ __launch_bounds__(256) void ew_gates(
    const float* __restrict__ h, const float* __restrict__ x,
    const float* __restrict__ mzh, const float* __restrict__ mzx,
    const float* __restrict__ mrh, const float* __restrict__ mrx,
    const float* __restrict__ bz, const float* __restrict__ br,
    float* __restrict__ zout, float* __restrict__ uxr,
    unsigned short* __restrict__ rphbf, float* __restrict__ rphss) {
    __shared__ float red[256];
    const int row = blockIdx.x;
    const int t   = threadIdx.x;
    const size_t base = (size_t)row * 1024 + t * 4;

    float h4[4], x4[4], a4[4], b4[4], c4[4], d4[4];
    *(float4*)h4 = *(const float4*)&h[base];
    *(float4*)x4 = *(const float4*)&x[base];
    *(float4*)a4 = *(const float4*)&mzh[base];
    *(float4*)b4 = *(const float4*)&mzx[base];
    *(float4*)c4 = *(const float4*)&mrh[base];
    *(float4*)d4 = *(const float4*)&mrx[base];

    // full-row (1024-dim) norms for mob_mat_mul scaling
    float hh = block_sum256(dot4(h4, h4), red, t);
    float xx = block_sum256(dot4(x4, x4), red, t);
    float s1 = block_sum256(dot4(a4, a4), red, t);
    float s2 = block_sum256(dot4(b4, b4), red, t);
    float s3 = block_sum256(dot4(c4, c4), red, t);
    float s4 = block_sum256(dot4(d4, d4), red, t);

    const float hn = clamp01(sqrtf(hh));
    const float xn = clamp01(sqrtf(xx));
    const float ath = atanhf(hn);
    const float atx = atanhf(xn);

    float n1 = fmaxf(sqrtf(s1), EPS);
    float n2 = fmaxf(sqrtf(s2), EPS);
    float n3 = fmaxf(sqrtf(s3), EPS);
    float n4 = fmaxf(sqrtf(s4), EPS);
    const float sc_zh = tanhf(n1 / hn * ath) / n1;
    const float sc_zx = tanhf(n2 / xn * atx) / n2;
    const float sc_rh = tanhf(n3 / hn * ath) / n3;
    const float sc_rx = tanhf(n4 / xn * atx) / n4;

    float bz4[4], br4[4];
    *(float4*)bz4 = *(const float4*)&bz[t * 4];
    *(float4*)br4 = *(const float4*)&br[t * 4];

    // z gate
    float wz[4], uz[4], t1[4], t2[4], lm[4], z4[4];
#pragma unroll
    for (int i = 0; i < 4; ++i) { wz[i] = sc_zh * a4[i]; uz[i] = sc_zx * b4[i]; }
    mob_add4(wz, uz, t1);
    mob_add4(t1, bz4, t2);
    logmap4(t2, lm);
#pragma unroll
    for (int i = 0; i < 4; ++i) z4[i] = 1.0f / (1.0f + expf(-lm[i]));
    *(float4*)&zout[base] = *(float4*)z4;

    // r gate
    float wr[4], ur[4], r4[4];
#pragma unroll
    for (int i = 0; i < 4; ++i) { wr[i] = sc_rh * c4[i]; ur[i] = sc_rx * d4[i]; }
    mob_add4(wr, ur, t1);
    mob_add4(t1, br4, t2);
    logmap4(t2, lm);
#pragma unroll
    for (int i = 0; i < 4; ++i) r4[i] = 1.0f / (1.0f + expf(-lm[i]));

    // r (.) h  (per-ball Mobius pointwise product) -> bf16 for the next GEMM
    float rph[4];
    mob_pp4(h4, r4, rph);
    rphbf[base + 0] = f2bf(rph[0]);
    rphbf[base + 1] = f2bf(rph[1]);
    rphbf[base + 2] = f2bf(rph[2]);
    rphbf[base + 3] = f2bf(rph[3]);
    float rs = block_sum256(dot4(rph, rph), red, t);
    if (t == 0) rphss[row] = rs;

    // scaled Ur@x (reused by h_tilde transition, per reference)
#pragma unroll
    for (int i = 0; i < 4; ++i) ur[i] = sc_rx * d4[i];
    *(float4*)&uxr[base] = *(float4*)ur;
}

// ---------------------------------------------------------------------------
// Elementwise phase 2: h_tilde, gated update, final Mobius add -> new_h
// ---------------------------------------------------------------------------
__global__ __launch_bounds__(256) void ew_update(
    const float* __restrict__ h, const float* __restrict__ z,
    const float* __restrict__ uxr, const float* __restrict__ mh,
    const float* __restrict__ bh, const float* __restrict__ rphss,
    float* __restrict__ out) {
    __shared__ float red[256];
    const int row = blockIdx.x;
    const int t   = threadIdx.x;
    const size_t base = (size_t)row * 1024 + t * 4;

    float h4[4], z4[4], u4[4], m4[4], bh4[4];
    *(float4*)h4  = *(const float4*)&h[base];
    *(float4*)z4  = *(const float4*)&z[base];
    *(float4*)u4  = *(const float4*)&uxr[base];
    *(float4*)m4  = *(const float4*)&mh[base];
    *(float4*)bh4 = *(const float4*)&bh[t * 4];

    float mss = block_sum256(dot4(m4, m4), red, t);
    float rpn = clamp01(sqrtf(rphss[row]));
    float mhn = fmaxf(sqrtf(mss), EPS);
    float sc  = tanhf(mhn / rpn * atanhf(rpn)) / mhn;

    float wh[4], t1[4], ht[4], nh[4], mi[4], pr[4], o4[4];
#pragma unroll
    for (int i = 0; i < 4; ++i) wh[i] = sc * m4[i];
    mob_add4(wh, u4, t1);
    mob_add4(t1, bh4, ht);
#pragma unroll
    for (int i = 0; i < 4; ++i) nh[i] = -h4[i];
    mob_add4(nh, ht, mi);
    mob_pp4(mi, z4, pr);
    mob_add4(h4, pr, o4);
    *(float4*)&out[base] = *(float4*)o4;
}

// ---------------------------------------------------------------------------
// Host launcher
// ---------------------------------------------------------------------------
extern "C" void kernel_launch(void* const* d_in, const int* in_sizes, int n_in,
                              void* d_out, int out_size, void* d_ws, size_t ws_size,
                              hipStream_t stream) {
    const float* hyp_x  = (const float*)d_in[0];
    const float* hidden = (const float*)d_in[1];
    const float* w_z    = (const float*)d_in[2];
    const float* w_r    = (const float*)d_in[3];
    const float* w_h    = (const float*)d_in[4];
    const float* u_z    = (const float*)d_in[5];
    const float* u_r    = (const float*)d_in[6];
    // d_in[7] (u_h) intentionally unused: reference uses u_r for h_tilde
    const float* b_z    = (const float*)d_in[8];
    const float* b_r    = (const float*)d_in[9];
    const float* b_h    = (const float*)d_in[10];
    float* out = (float*)d_out;

    const int ROW = 1024;                    // H*D = I*D
    const int B   = in_sizes[1] / ROW;       // batch (16384)
    const size_t act = (size_t)B * ROW;      // activation elements
    const size_t wel = (size_t)ROW * ROW;    // weight elements

    // Workspace carve-out (256B aligned)
    char* ws = (char*)d_ws;
    size_t off = 0;
    auto carve = [&](size_t bytes) -> char* {
        char* p = ws + off;
        off += (bytes + 255) & ~(size_t)255;
        return p;
    };
    unsigned short* Hbf = (unsigned short*)carve(act * 2);
    unsigned short* Xbf = (unsigned short*)carve(act * 2);
    unsigned short* Wzb = (unsigned short*)carve(wel * 2);   // transposed (N,K)
    unsigned short* Wrb = (unsigned short*)carve(wel * 2);
    unsigned short* Whb = (unsigned short*)carve(wel * 2);
    unsigned short* Uzb = (unsigned short*)carve(wel * 2);
    unsigned short* Urb = (unsigned short*)carve(wel * 2);
    float* Mzh = (float*)carve(act * 4);
    float* Mzx = (float*)carve(act * 4);
    float* Mrh = (float*)carve(act * 4);
    float* Mrx = (float*)carve(act * 4);
    float* Zb  = (float*)carve(act * 4);
    float* Uxr = (float*)carve(act * 4);
    unsigned short* RPH = (unsigned short*)carve(act * 2);
    float* RPHSS = (float*)carve((size_t)B * 4);
    float* Mh = Mzh;  // safe alias: Mzh fully consumed by ew_gates before reuse

    // 1) f32 -> bf16 conversions (weights transposed once -> contiguous tiles)
    const int cvb = 256;
    cvt_bf16<<<dim3((unsigned)((act / 4 + cvb - 1) / cvb)), cvb, 0, stream>>>(hidden, Hbf, (int)act);
    cvt_bf16<<<dim3((unsigned)((act / 4 + cvb - 1) / cvb)), cvb, 0, stream>>>(hyp_x, Xbf, (int)act);
    dim3 tg(ROW / 32, ROW / 32);
    cvt_bf16_T<<<tg, cvb, 0, stream>>>(w_z, Wzb, ROW, ROW);
    cvt_bf16_T<<<tg, cvb, 0, stream>>>(w_r, Wrb, ROW, ROW);
    cvt_bf16_T<<<tg, cvb, 0, stream>>>(w_h, Whb, ROW, ROW);
    cvt_bf16_T<<<tg, cvb, 0, stream>>>(u_z, Uzb, ROW, ROW);
    cvt_bf16_T<<<tg, cvb, 0, stream>>>(u_r, Urb, ROW, ROW);

    // 2) phase-1 GEMMs (WMMA bf16 -> f32, async double-buffered tiles)
    dim3 gg(ROW / TN, B / TM);
    gemm_bf16<<<gg, 256, 0, stream>>>(Hbf, Wzb, Mzh, B, ROW, ROW);
    gemm_bf16<<<gg, 256, 0, stream>>>(Xbf, Uzb, Mzx, B, ROW, ROW);
    gemm_bf16<<<gg, 256, 0, stream>>>(Hbf, Wrb, Mrh, B, ROW, ROW);
    gemm_bf16<<<gg, 256, 0, stream>>>(Xbf, Urb, Mrx, B, ROW, ROW);

    // 3) gates + r.h + reusable Ur@x
    ew_gates<<<dim3(B), 256, 0, stream>>>(hidden, hyp_x, Mzh, Mzx, Mrh, Mrx,
                                          b_z, b_r, Zb, Uxr, RPH, RPHSS);

    // 4) phase-2 GEMM: (r.h) @ Wh
    gemm_bf16<<<gg, 256, 0, stream>>>(RPH, Whb, Mh, B, ROW, ROW);

    // 5) h_tilde + gated update -> new_h
    ew_update<<<dim3(B), 256, 0, stream>>>(hidden, Zb, Uxr, Mh, b_h, RPHSS, out);
}
